// BertSelfAttention1D_41979010351408
// MI455X (gfx1250) — matrix operand order repaired and checked by reading, not compiled, not run on hardware
//
#include <hip/hip_runtime.h>
#include <hip/hip_bf16.h>

// ---------------------------------------------------------------------------
// BERT self-attention block on gfx1250 (MI455X), bf16 WMMA + f32 accumulate,
// async global->LDS staging, flash-attention streaming softmax.
// ---------------------------------------------------------------------------
typedef __bf16 bf16;
typedef __attribute__((ext_vector_type(16))) __bf16 v16bf;
typedef __attribute__((ext_vector_type(8)))  float  v8f;

#define HID   1024
#define HEADS 16
#define HDIM  64
#define NB    2
#define SEQ   2048
#define ROWS  (NB * SEQ)      // 4096
#define QKVN  (3 * HID)       // 3072
#define LN_EPS 1e-5f

#define ASYNC_LDS 1           // CDNA5 GLOBAL_LOAD_ASYNC_TO_LDS path

// 16 bf16 (32B) fragment viewed as two 16B chunks for LDS/global b128 loads.
union FragB { v16bf v; uint4 q[2]; };

static __device__ __forceinline__ v8f wmma_bf16(const v16bf& a, const v16bf& b, v8f c) {
  return __builtin_amdgcn_wmma_f32_16x16x32_bf16(false, a, false, b, (short)0, c,
                                                 false, false);
}

// Async copy 16B: per-lane LDS[lds_off] <- MEM[gaddr]; tracked by ASYNCcnt.
static __device__ __forceinline__ void async_ld_b128(unsigned lds_off, const void* gptr) {
  unsigned long long ga = (unsigned long long)gptr;
  asm volatile("global_load_async_to_lds_b128 %0, %1, off"
               :: "v"(lds_off), "v"(ga) : "memory");
}
static __device__ __forceinline__ void wait_async0() {
  asm volatile("s_wait_asynccnt 0x0" ::: "memory");
}
static __device__ __forceinline__ unsigned lds_off_of(const void* p) {
  return (unsigned)(size_t)p;   // generic LDS pointer: low 32 bits = LDS byte offset
}

// Max-reduction butterfly across each 16-lane row (VALU permlane if available).
static __device__ __forceinline__ float rowmax16(float x) {
#if __has_builtin(__builtin_amdgcn_permlane16)
  union { float f; unsigned u; } a, b;
  a.f = x;
  b.u = (unsigned)__builtin_amdgcn_permlane16((int)a.u, (int)a.u, 0x67452301, 0xEFCDAB89, false, false); // xor 1
  a.f = fmaxf(a.f, b.f);
  b.u = (unsigned)__builtin_amdgcn_permlane16((int)a.u, (int)a.u, 0x54761032, 0xDCFE98BA, false, false); // xor 2
  a.f = fmaxf(a.f, b.f);
  b.u = (unsigned)__builtin_amdgcn_permlane16((int)a.u, (int)a.u, 0x32107654, 0xBA98FEDC, false, false); // xor 4
  a.f = fmaxf(a.f, b.f);
  b.u = (unsigned)__builtin_amdgcn_permlane16((int)a.u, (int)a.u, 0xFEDCBA98, 0x76543210, false, false); // xor 8
  return fmaxf(a.f, b.f);
#else
  for (int off = 1; off < 16; off <<= 1) x = fmaxf(x, __shfl_xor(x, off, 32));
  return x;
#endif
}

// ---------------------------------------------------------------------------
// Elementwise fp32 -> bf16 convert
// ---------------------------------------------------------------------------
__global__ void __launch_bounds__(256)
cvt_f32_bf16(const float* __restrict__ in, bf16* __restrict__ out, int n) {
  int i = blockIdx.x * 256 + threadIdx.x;
  if (i < n) out[i] = (bf16)in[i];
}

// ---------------------------------------------------------------------------
// Transpose + convert: in[R][C] f32  ->  out[C][R] bf16   (32x32 LDS tiles)
// ---------------------------------------------------------------------------
__global__ void __launch_bounds__(256)
transpose_cvt(const float* __restrict__ in, bf16* __restrict__ out, int R, int C) {
  __shared__ float t[32][33];
  const int bx = blockIdx.x * 32;
  const int by = blockIdx.y * 32;
  const int tx = threadIdx.x & 31, ty = threadIdx.x >> 5;
  for (int i = 0; i < 32; i += 8)
    t[ty + i][tx] = in[(size_t)(by + ty + i) * C + bx + tx];
  __syncthreads();
  for (int i = 0; i < 32; i += 8)
    out[(size_t)(bx + ty + i) * R + by + tx] = (bf16)t[tx][ty + i];
}

// ---------------------------------------------------------------------------
// Tiled WMMA GEMM:  C[M][N] = A[M][K](bf16) * B (bf16, stored as Bt[N][K]) + bias
// Block tile 128x128, BK=32, 256 threads = 8 waves (4 x 2), wave tile 32x64.
// SCALE_Q: multiply Q columns (n%192 < 64) by 1/sqrt(HDIM) in the epilogue.
// ---------------------------------------------------------------------------
template <bool OUT_BF16, bool SCALE_Q>
__global__ void __launch_bounds__(256)
gemm_bf16_wmma(const bf16* __restrict__ A, const bf16* __restrict__ Bt,
               const float* __restrict__ bias, void* __restrict__ Cout,
               int N, int K) {
  __shared__ __attribute__((aligned(16))) bf16 As[128 * 40];  // 80B rows
  __shared__ __attribute__((aligned(16))) bf16 Bs[128 * 40];

  const int tid  = threadIdx.x;
  const int wave = tid >> 5, lane = tid & 31;
  const int lr   = lane & 15, hi = (lane >> 4) & 1;
  const int wm   = wave >> 1, wn = wave & 1;
  const int m0   = blockIdx.y * 128, n0 = blockIdx.x * 128;

  v8f acc[2][4];
  for (int i = 0; i < 2; ++i)
    for (int j = 0; j < 4; ++j) acc[i][j] = (v8f){};

  for (int kb = 0; kb < K; kb += 32) {
    for (int i = 0; i < 2; ++i) {
      const int chunk = tid + i * 256;             // 0..511
      const int row = chunk >> 2;                  // 0..127
      const int c8  = (chunk & 3) << 3;            // 0,8,16,24
#if ASYNC_LDS
      async_ld_b128(lds_off_of(&As[row * 40 + c8]),
                    &A[(size_t)(m0 + row) * K + kb + c8]);
      async_ld_b128(lds_off_of(&Bs[row * 40 + c8]),
                    &Bt[(size_t)(n0 + row) * K + kb + c8]);
#else
      *(uint4*)&As[row * 40 + c8] = *(const uint4*)&A [(size_t)(m0 + row) * K + kb + c8];
      *(uint4*)&Bs[row * 40 + c8] = *(const uint4*)&Bt[(size_t)(n0 + row) * K + kb + c8];
#endif
    }
#if ASYNC_LDS
    wait_async0();
#endif
    __syncthreads();

    FragB a[2], b[4];
    for (int i = 0; i < 2; ++i) {
      // A-frag: lane-group 0 -> K {0..7,16..23}; group 1 -> K {8..15,24..31}
      const bf16* p = &As[(wm * 32 + i * 16 + lr) * 40];
      a[i].q[0] = *(const uint4*)(p + hi * 8);
      a[i].q[1] = *(const uint4*)(p + 16 + hi * 8);
    }
    for (int j = 0; j < 4; ++j) {
      // B-frag: lane-group 0 -> K {0..7,16..23}; group 1 -> K {8..15,24..31}
      const bf16* p = &Bs[(wn * 64 + j * 16 + lr) * 40];
      b[j].q[0] = *(const uint4*)(p + hi * 8);
      b[j].q[1] = *(const uint4*)(p + 16 + hi * 8);
    }
    for (int i = 0; i < 2; ++i)
      for (int j = 0; j < 4; ++j)
        acc[i][j] = wmma_bf16(a[i].v, b[j].v, acc[i][j]);
    __syncthreads();
  }

  // Epilogue: C-tile layout  M = r + 8*hi, N = lr
  for (int j = 0; j < 4; ++j) {
    const int n  = n0 + wn * 64 + j * 16 + lr;
    const float bn = bias[n];
    const float sc = (SCALE_Q && ((n % 192) < HDIM)) ? 0.125f : 1.0f;
    for (int i = 0; i < 2; ++i) {
      const int mbase = m0 + wm * 32 + i * 16 + hi * 8;
      for (int r = 0; r < 8; ++r) {
        const float v = (acc[i][j][r] + bn) * sc;
        const size_t idx = (size_t)(mbase + r) * N + n;
        if (OUT_BF16) ((bf16*)Cout)[idx] = (bf16)v;
        else          ((float*)Cout)[idx] = v;
      }
    }
  }
}

// ---------------------------------------------------------------------------
// Flash attention, no mask.  qkv columns per head h: [h*192,+64)=Q (pre-scaled
// by 1/8), [+64..+127]=K, [+128..+191]=V.  Grid: (SEQ/128, HEADS, NB).
// 8 waves; each wave owns 16 query rows, streams keys in 64-wide tiles.
// QK^T, P@V and rowsum(P)=P@ones all on WMMA.
// ---------------------------------------------------------------------------
__global__ void __launch_bounds__(256)
attn_wmma(const bf16* __restrict__ qkv, bf16* __restrict__ ctx) {
  __shared__ __attribute__((aligned(16))) bf16 Ks[64 * 72];       // K tile rows
  __shared__ __attribute__((aligned(16))) bf16 Vt[64 * 72];       // V tile transposed
  __shared__ __attribute__((aligned(16))) bf16 Ps[8 * 16 * 72];   // per-wave P staging

  const int tid  = threadIdx.x;
  const int wave = tid >> 5, lane = tid & 31;
  const int lr   = lane & 15, hi = (lane >> 4) & 1;
  const int h = blockIdx.y, b = blockIdx.z;
  const int q0 = blockIdx.x * 128 + wave * 16;

  const size_t rs = QKVN;
  const bf16* Qp = qkv + ((size_t)(b * SEQ + q0)) * rs + h * (3 * HDIM);
  const bf16* Kp = qkv + ((size_t)(b * SEQ)) * rs + h * (3 * HDIM) + HDIM;
  const bf16* Vp = Kp + HDIM;

  // Q A-fragments (d = 0..31, 32..63), reused for all key tiles.
  FragB qa[2];
  {
    const bf16* p = Qp + (size_t)lr * rs;
    qa[0].q[0] = *(const uint4*)(p +      hi * 8);
    qa[0].q[1] = *(const uint4*)(p + 16 + hi * 8);
    qa[1].q[0] = *(const uint4*)(p + 32 + hi * 8);
    qa[1].q[1] = *(const uint4*)(p + 48 + hi * 8);
  }

  // Ones B-fragment for rowsum(P) = P @ ones.
  FragB ones;
  ones.q[0] = make_uint4(0x3F803F80u, 0x3F803F80u, 0x3F803F80u, 0x3F803F80u);
  ones.q[1] = ones.q[0];

  float m_i[8];
  v8f O[4] = {};
  v8f L = {};                    // running softmax denominator (all cols equal)
  for (int r = 0; r < 8; ++r) m_i[r] = -1e30f;

  for (int kt = 0; kt < SEQ / 64; ++kt) {
    const int kbase = kt * 64;
    for (int i = 0; i < 2; ++i) {
      const int chunk = tid + i * 256;            // 0..511
      const int row = chunk >> 3;                 // key 0..63
      const int c8  = (chunk & 7) << 3;           // d 0..56
#if ASYNC_LDS
      async_ld_b128(lds_off_of(&Ks[row * 72 + c8]),
                    &Kp[(size_t)(kbase + row) * rs + c8]);
#else
      *(uint4*)&Ks[row * 72 + c8] =
          *(const uint4*)&Kp[(size_t)(kbase + row) * rs + c8];
#endif
      // V^T staged with a scalar scatter (transpose).
      uint4 vv = *(const uint4*)&Vp[(size_t)(kbase + row) * rs + c8];
      const bf16* ve = (const bf16*)&vv;
      for (int e = 0; e < 8; ++e) Vt[(c8 + e) * 72 + row] = ve[e];
    }
#if ASYNC_LDS
    wait_async0();
#endif
    __syncthreads();

    // S = Q K^T : 16 queries x 64 keys (Q pre-scaled by 1/sqrt(d)).
    v8f s[4];
    for (int j = 0; j < 4; ++j) {
      FragB b0, b1;
      const bf16* p = &Ks[(j * 16 + lr) * 72];
      b0.q[0] = *(const uint4*)(p + hi * 8);
      b0.q[1] = *(const uint4*)(p + 16 + hi * 8);
      b1.q[0] = *(const uint4*)(p + 32 + hi * 8);
      b1.q[1] = *(const uint4*)(p + 48 + hi * 8);
      v8f z = {};
      z = wmma_bf16(qa[0].v, b0.v, z);
      z = wmma_bf16(qa[1].v, b1.v, z);
      s[j] = z;
    }

    // Streaming softmax: row max via VALU permlane butterfly; exp in registers.
    float mnew[8], alpha[8];
    for (int r = 0; r < 8; ++r) {
      float mx = fmaxf(fmaxf(s[0][r], s[1][r]), fmaxf(s[2][r], s[3][r]));
      mx = rowmax16(mx);
      mnew[r]  = fmaxf(m_i[r], mx);
      alpha[r] = __expf(m_i[r] - mnew[r]);
      m_i[r]   = mnew[r];
    }
    for (int j = 0; j < 4; ++j)
      for (int r = 0; r < 8; ++r)
        s[j][r] = __expf(s[j][r] - mnew[r]);
    for (int jd = 0; jd < 4; ++jd)
      for (int r = 0; r < 8; ++r) O[jd][r] *= alpha[r];
    for (int r = 0; r < 8; ++r) L[r] *= alpha[r];

    // C-layout P -> A-layout via per-wave LDS staging (row = r+8*hi, col = key).
    bf16* myP = &Ps[wave * 16 * 72];
    for (int j = 0; j < 4; ++j)
      for (int r = 0; r < 8; ++r)
        myP[(r + hi * 8) * 72 + j * 16 + lr] = (bf16)s[j][r];

    FragB pa[2];
    {
      const bf16* p = &myP[lr * 72];
      pa[0].q[0] = *(const uint4*)(p +      hi * 8);
      pa[0].q[1] = *(const uint4*)(p + 16 + hi * 8);
      pa[1].q[0] = *(const uint4*)(p + 32 + hi * 8);
      pa[1].q[1] = *(const uint4*)(p + 48 + hi * 8);
    }

    // Running denominator: L += rowsum(P) = P @ ones (matrix pipe, no shuffles).
    L = wmma_bf16(pa[0].v, ones.v, L);
    L = wmma_bf16(pa[1].v, ones.v, L);

    // O += P @ V   (B operand = V^T rows)
    for (int jd = 0; jd < 4; ++jd) {
      FragB b0, b1;
      const bf16* p = &Vt[(jd * 16 + lr) * 72];
      b0.q[0] = *(const uint4*)(p + hi * 8);
      b0.q[1] = *(const uint4*)(p + 16 + hi * 8);
      b1.q[0] = *(const uint4*)(p + 32 + hi * 8);
      b1.q[1] = *(const uint4*)(p + 48 + hi * 8);
      O[jd] = wmma_bf16(pa[0].v, b0.v, O[jd]);
      O[jd] = wmma_bf16(pa[1].v, b1.v, O[jd]);
    }
    __syncthreads();
  }

  // ctx[b][q][h*64 + d] = O / L
  for (int jd = 0; jd < 4; ++jd) {
    const int d = h * HDIM + jd * 16 + lr;
    for (int r = 0; r < 8; ++r) {
      const int q = q0 + hi * 8 + r;
      ctx[((size_t)(b * SEQ + q)) * HID + d] = (bf16)(O[jd][r] / L[r]);
    }
  }
}

// ---------------------------------------------------------------------------
// x = y + resid;  out = LN(x) * gamma + beta.  One block per row of 1024.
// ---------------------------------------------------------------------------
__global__ void __launch_bounds__(256)
add_ln(const float* __restrict__ y, const float* __restrict__ resid,
       const float* __restrict__ gamma, const float* __restrict__ beta,
       float* __restrict__ out) {
  const int row = blockIdx.x, tid = threadIdx.x;
  const float* yr = y + (size_t)row * HID;
  const float* rr = resid + (size_t)row * HID;
  float x[4], s = 0.f, s2 = 0.f;
  for (int i = 0; i < 4; ++i) {
    const int c = tid + i * 256;
    x[i] = yr[c] + rr[c];
    s += x[i]; s2 += x[i] * x[i];
  }
  for (int off = 1; off < 32; off <<= 1) {
    s += __shfl_xor(s, off, 32); s2 += __shfl_xor(s2, off, 32);
  }
  __shared__ float ws[8], ws2[8];
  const int wave = tid >> 5, lane = tid & 31;
  if (lane == 0) { ws[wave] = s; ws2[wave] = s2; }
  __syncthreads();
  if (wave == 0) {
    float a  = (lane < 8) ? ws[lane]  : 0.f;
    float a2 = (lane < 8) ? ws2[lane] : 0.f;
    for (int off = 1; off < 8; off <<= 1) {
      a += __shfl_xor(a, off, 32); a2 += __shfl_xor(a2, off, 32);
    }
    if (lane == 0) { ws[0] = a; ws2[0] = a2; }
  }
  __syncthreads();
  const float mu   = ws[0] * (1.0f / HID);
  const float var  = ws2[0] * (1.0f / HID) - mu * mu;
  const float rstd = rsqrtf(var + LN_EPS);
  float* outr = out + (size_t)row * HID;
  for (int i = 0; i < 4; ++i) {
    const int c = tid + i * 256;
    outr[c] = (x[i] - mu) * rstd * gamma[c] + beta[c];
  }
}

// ---------------------------------------------------------------------------
extern "C" void kernel_launch(void* const* d_in, const int* in_sizes, int n_in,
                              void* d_out, int out_size, void* d_ws, size_t ws_size,
                              hipStream_t stream) {
  const float* hidden  = (const float*)d_in[0];
  const float* w_qkv   = (const float*)d_in[1];
  const float* b_qkv   = (const float*)d_in[2];
  const float* w_dense = (const float*)d_in[3];
  const float* b_dense = (const float*)d_in[4];
  const float* gamma   = (const float*)d_in[5];
  const float* beta    = (const float*)d_in[6];
  float* out = (float*)d_out;

  char* p = (char*)d_ws;
  bf16*  Xb    = (bf16*)p;  p += (size_t)ROWS * HID  * sizeof(bf16);
  bf16*  WqkvT = (bf16*)p;  p += (size_t)QKVN * HID  * sizeof(bf16);
  bf16*  WdT   = (bf16*)p;  p += (size_t)HID  * HID  * sizeof(bf16);
  bf16*  QKVb  = (bf16*)p;  p += (size_t)ROWS * QKVN * sizeof(bf16);
  bf16*  CTXb  = (bf16*)p;  p += (size_t)ROWS * HID  * sizeof(bf16);
  float* Y     = (float*)p; p += (size_t)ROWS * HID  * sizeof(float);
  (void)ws_size; (void)in_sizes; (void)n_in; (void)out_size;

  {
    const int n = ROWS * HID;
    cvt_f32_bf16<<<(n + 255) / 256, 256, 0, stream>>>(hidden, Xb, n);
  }
  transpose_cvt<<<dim3(QKVN / 32, HID / 32), 256, 0, stream>>>(w_qkv, WqkvT, HID, QKVN);
  transpose_cvt<<<dim3(HID / 32, HID / 32), 256, 0, stream>>>(w_dense, WdT, HID, HID);
  // QKV projection (+bias), Q columns pre-scaled by 1/sqrt(HDIM), bf16 out
  gemm_bf16_wmma<true, true><<<dim3(QKVN / 128, ROWS / 128), 256, 0, stream>>>(
      Xb, WqkvT, b_qkv, QKVb, QKVN, HID);
  attn_wmma<<<dim3(SEQ / 128, HEADS, NB), 256, 0, stream>>>(QKVb, CTXb);
  gemm_bf16_wmma<false, false><<<dim3(HID / 128, ROWS / 128), 256, 0, stream>>>(
      CTXb, WdT, b_dense, Y, HID, HID);
  add_ln<<<ROWS, 256, 0, stream>>>(Y, hidden, gamma, beta, out);
}
